// UnifiedMoE_65420941852891
// MI455X (gfx1250) — matrix operand order repaired and checked by reading, not compiled, not run on hardware
//
#include <hip/hip_runtime.h>
#include <hip/hip_bf16.h>

// ---- problem sizes (fixed by reference) ----
#define T_ 512
#define D_ 1024
#define E_ 64
#define I_ 512
#define K_TOP 4
#define TILE_M 16
#define XPAD 8   // LDS row padding (elements) to break bank alignment

typedef __attribute__((ext_vector_type(16))) _Float16 v16h;
typedef __attribute__((ext_vector_type(8)))  float    v8f;

// ---------------------------------------------------------------------------
// A-fragment (16x32 f16, M x K) from an LDS row.
// lane = 16*half + m ; element j -> K = k0 + 16*half + j
// i.e. two contiguous 8-element (16B) runs.
// ---------------------------------------------------------------------------
__device__ inline v16h make_a_frag(const _Float16* row_base, int k0, int half) {
    union { v16h v; uint4 u[2]; } f;
    f.u[0] = *reinterpret_cast<const uint4*>(row_base + k0 + 16 * half);
    f.u[1] = *reinterpret_cast<const uint4*>(row_base + k0 + 16 * half + 8);
    return f.v;
}

// ---------------------------------------------------------------------------
// B-fragment (32x16 f16, K x N) from global fp32 weights stored with K
// contiguous per output column.  lane = 16*half + n ; element j -> K = k0 + 16*half + j
// => 16 consecutive floats, converted to f16.
// ---------------------------------------------------------------------------
__device__ inline v16h make_b_frag(const float* col_base, int k0, int half) {
    const float4* p = reinterpret_cast<const float4*>(col_base + k0 + 16 * half);
    v16h r;
#pragma unroll
    for (int i = 0; i < 4; ++i) {
        float4 f4 = p[i];
        r[4 * i + 0] = (_Float16)f4.x;
        r[4 * i + 1] = (_Float16)f4.y;
        r[4 * i + 2] = (_Float16)f4.z;
        r[4 * i + 3] = (_Float16)f4.w;
    }
    return r;
}

// ---------------------------------------------------------------------------
// zero counts + output
// ---------------------------------------------------------------------------
__global__ void moe_zero_kernel(float* __restrict__ out, int n_out,
                                int* __restrict__ counts) {
    int i = blockIdx.x * blockDim.x + threadIdx.x;
    if (i < n_out) out[i] = 0.0f;
    if (i < E_) counts[i] = 0;
}

// ---------------------------------------------------------------------------
// Router: logits -> softmax -> top-4 (first-index tie-break) -> renormalize.
// One block per token, 64 threads (one per expert).
// ---------------------------------------------------------------------------
__global__ void moe_router_kernel(const float* __restrict__ x,
                                  const float* __restrict__ gate_w,
                                  int* __restrict__ counts,
                                  int* __restrict__ tok_ids,
                                  int* __restrict__ tok_slot,
                                  float* __restrict__ tok_w) {
    int t = blockIdx.x;
    int e = threadIdx.x;  // 0..63

    __shared__ float probs[E_];
    __shared__ float red[E_];
    __shared__ int   sel_id[K_TOP];
    __shared__ float sel_w[K_TOP];

    const float4* xr = reinterpret_cast<const float4*>(x + (size_t)t * D_);
    const float4* gr = reinterpret_cast<const float4*>(gate_w + (size_t)e * D_);
    float acc = 0.0f;
    for (int i = 0; i < D_ / 4; ++i) {
        float4 a = xr[i], b = gr[i];
        acc += a.x * b.x + a.y * b.y + a.z * b.z + a.w * b.w;
    }
    red[e] = acc;
    __syncthreads();
    for (int s = 32; s > 0; s >>= 1) {
        if (e < s) red[e] = fmaxf(red[e], red[e + s]);
        __syncthreads();
    }
    float mx = red[0];
    __syncthreads();
    float p = __expf(acc - mx);
    red[e] = p;
    __syncthreads();
    for (int s = 32; s > 0; s >>= 1) {
        if (e < s) red[e] += red[e + s];
        __syncthreads();
    }
    float denom = red[0];
    __syncthreads();
    probs[e] = p / denom;
    __syncthreads();

    if (e == 0) {
        float wsum = 0.0f;
        for (int k = 0; k < K_TOP; ++k) {
            int best = 0;
            float bv = -1.0f;
            for (int j = 0; j < E_; ++j) {
                bool taken = false;
                for (int m = 0; m < k; ++m) taken |= (sel_id[m] == j);
                float v = probs[j];
                if (!taken && v > bv) { bv = v; best = j; }
            }
            sel_id[k] = best;
            sel_w[k]  = bv;
            wsum += bv;
        }
        float inv = 1.0f / wsum;
        for (int k = 0; k < K_TOP; ++k) sel_w[k] *= inv;
    }
    __syncthreads();

    if (e < K_TOP) {
        int ex  = sel_id[e];
        int pos = atomicAdd(&counts[ex], 1);
        tok_ids [ex * T_ + pos] = t;
        tok_slot[ex * T_ + pos] = e;      // which of the top-4 slots
        tok_w   [ex * T_ + pos] = sel_w[e];
    }
}

// ---------------------------------------------------------------------------
// Expert FFN: one block = (expert, 16-token tile), 8 waves (256 threads).
// GEMM1 (gate+up, K=D), SiLU*up -> LDS, GEMM2 (down, K=INTER), weighted write.
// ---------------------------------------------------------------------------
__global__ void __launch_bounds__(256)
moe_expert_kernel(const float* __restrict__ x,
                  const float* __restrict__ w_gate,
                  const float* __restrict__ w_up,
                  const float* __restrict__ w_down,
                  const int*   __restrict__ counts,
                  const int*   __restrict__ tok_ids,
                  const int*   __restrict__ tok_slot,
                  const float* __restrict__ tok_w,
                  float* __restrict__ y_part,   // [T][K_TOP][D] or nullptr
                  float* __restrict__ out) {    // [T][D] (atomic fallback)
    const int e    = blockIdx.x;
    const int tile = blockIdx.y;
    const int cnt  = counts[e];
    if (tile * TILE_M >= cnt) return;

    __shared__ __align__(16) _Float16 Xs[TILE_M][D_ + XPAD];
    __shared__ __align__(16) _Float16 Hs[TILE_M][I_ + XPAD];
    __shared__ int   tid_s[TILE_M];
    __shared__ int   tsl_s[TILE_M];
    __shared__ float tw_s[TILE_M];

    const int t    = threadIdx.x;    // 0..255
    const int wave = t >> 5;         // 0..7
    const int lane = t & 31;
    const int half = lane >> 4;      // 0/1
    const int ln16 = lane & 15;      // 0..15

    if (t < TILE_M) {
        int idx = tile * TILE_M + t;
        if (idx < cnt) {
            tid_s[t] = tok_ids [e * T_ + idx];
            tsl_s[t] = tok_slot[e * T_ + idx];
            tw_s[t]  = tok_w   [e * T_ + idx];
        } else {
            tid_s[t] = 0; tsl_s[t] = 0; tw_s[t] = 0.0f;
        }
    }
    __syncthreads();

    // ---- stage X tile (16 x 1024) into LDS as f16, zero padded rows ----
    {
        int r  = t >> 4;       // 0..15
        int cg = t & 15;       // 0..15
        bool valid = (tile * TILE_M + r) < cnt;
        const float4* src = reinterpret_cast<const float4*>(x + (size_t)tid_s[r] * D_);
        for (int c4 = cg; c4 < D_ / 4; c4 += 16) {
            float4 v;
            if (valid) v = src[c4];
            else       v = make_float4(0.0f, 0.0f, 0.0f, 0.0f);
            int c = c4 * 4;
            Xs[r][c + 0] = (_Float16)v.x;
            Xs[r][c + 1] = (_Float16)v.y;
            Xs[r][c + 2] = (_Float16)v.z;
            Xs[r][c + 3] = (_Float16)v.w;
        }
    }
    __syncthreads();

    // ---- GEMM1: [16 x D] x [D x 64]  (gate & up), this wave's 64 columns ----
    const float* wg = w_gate + (size_t)e * I_ * D_;
    const float* wu = w_up   + (size_t)e * I_ * D_;

    v8f accg[4], accu[4];
    {
        v8f z = {};
#pragma unroll
        for (int nt = 0; nt < 4; ++nt) { accg[nt] = z; accu[nt] = z; }
    }
    const _Float16* a_row = &Xs[ln16][0];
    for (int k0 = 0; k0 < D_; k0 += 32) {
        v16h a = make_a_frag(a_row, k0, half);
#pragma unroll
        for (int nt = 0; nt < 4; ++nt) {
            int col = wave * 64 + nt * 16 + ln16;
            v16h bg = make_b_frag(wg + (size_t)col * D_, k0, half);
            v16h bu = make_b_frag(wu + (size_t)col * D_, k0, half);
            accg[nt] = __builtin_amdgcn_wmma_f32_16x16x32_f16(
                false, a, false, bg, (short)0, accg[nt], false, false);
            accu[nt] = __builtin_amdgcn_wmma_f32_16x16x32_f16(
                false, a, false, bu, (short)0, accu[nt], false, false);
        }
    }

    // ---- SiLU(g) * u -> Hs (f16) ----
#pragma unroll
    for (int nt = 0; nt < 4; ++nt) {
        int col = wave * 64 + nt * 16 + ln16;
#pragma unroll
        for (int r = 0; r < 8; ++r) {
            int row = half * 8 + r;
            float g = accg[nt][r];
            float u = accu[nt][r];
            float h = (g / (1.0f + __expf(-g))) * u;
            Hs[row][col] = (_Float16)h;
        }
    }
    __syncthreads();

    // ---- GEMM2: [16 x INTER] x [INTER x 128], this wave's 128 columns ----
    const float* wd = w_down + (size_t)e * D_ * I_;
    v8f accy[8];
    {
        v8f z = {};
#pragma unroll
        for (int nt = 0; nt < 8; ++nt) accy[nt] = z;
    }
    const _Float16* h_row = &Hs[ln16][0];
    for (int k0 = 0; k0 < I_; k0 += 32) {
        v16h a = make_a_frag(h_row, k0, half);
#pragma unroll
        for (int nt = 0; nt < 8; ++nt) {
            int col = wave * 128 + nt * 16 + ln16;
            v16h b = make_b_frag(wd + (size_t)col * I_, k0, half);
            accy[nt] = __builtin_amdgcn_wmma_f32_16x16x32_f16(
                false, a, false, b, (short)0, accy[nt], false, false);
        }
    }

    // ---- weighted writeback ----
#pragma unroll
    for (int nt = 0; nt < 8; ++nt) {
        int col = wave * 128 + nt * 16 + ln16;
#pragma unroll
        for (int r = 0; r < 8; ++r) {
            int row = half * 8 + r;
            if (tile * TILE_M + row < cnt) {
                float v = tw_s[row] * accy[nt][r];
                int token = tid_s[row];
                if (y_part) {
                    y_part[((size_t)token * K_TOP + tsl_s[row]) * D_ + col] = v;
                } else {
                    atomicAdd(&out[(size_t)token * D_ + col], v);
                }
            }
        }
    }
}

// ---------------------------------------------------------------------------
// Deterministic fixed-order combine of the 4 per-slot partials.
// ---------------------------------------------------------------------------
__global__ void moe_combine_kernel(const float* __restrict__ yp,
                                   float* __restrict__ out, int n) {
    int i = blockIdx.x * blockDim.x + threadIdx.x;
    if (i >= n) return;
    int tk = i / D_;
    int c  = i - tk * D_;
    const float* p = yp + ((size_t)tk * K_TOP) * D_ + c;
    out[i] = ((p[0] + p[(size_t)D_]) + p[2 * (size_t)D_]) + p[3 * (size_t)D_];
}

// ---------------------------------------------------------------------------
extern "C" void kernel_launch(void* const* d_in, const int* in_sizes, int n_in,
                              void* d_out, int out_size, void* d_ws, size_t ws_size,
                              hipStream_t stream) {
    (void)in_sizes; (void)n_in; (void)out_size;

    const float* x      = (const float*)d_in[0];
    const float* gate_w = (const float*)d_in[1];
    const float* w_gate = (const float*)d_in[2];
    const float* w_up   = (const float*)d_in[3];
    const float* w_down = (const float*)d_in[4];
    float* out = (float*)d_out;

    // workspace carve-up
    char* ws = (char*)d_ws;
    int*   counts   = (int*)ws;                                     // E ints (pad 256B)
    int*   tok_ids  = (int*)(ws + 256);                             // E*T ints
    int*   tok_slot = (int*)(ws + 256 + (size_t)E_ * T_ * 4);       // E*T ints
    float* tok_w    = (float*)(ws + 256 + 2ull * E_ * T_ * 4);      // E*T floats
    float* y_part   = (float*)(ws + 256 + 3ull * E_ * T_ * 4);      // T*K*D floats

    size_t need = 256 + 3ull * E_ * T_ * 4 + (size_t)T_ * K_TOP * D_ * 4;
    bool use_part = (ws_size >= need);

    const int n_out = T_ * D_;
    moe_zero_kernel<<<(n_out + 255) / 256, 256, 0, stream>>>(out, n_out, counts);

    moe_router_kernel<<<T_, E_, 0, stream>>>(x, gate_w, counts, tok_ids, tok_slot, tok_w);

    dim3 grid(E_, T_ / TILE_M);
    moe_expert_kernel<<<grid, 256, 0, stream>>>(x, w_gate, w_up, w_down,
                                                counts, tok_ids, tok_slot, tok_w,
                                                use_part ? y_part : nullptr, out);

    if (use_part) {
        moe_combine_kernel<<<(n_out + 255) / 256, 256, 0, stream>>>(y_part, out, n_out);
    }
}
